// NaiveDSA_72413148611124
// MI455X (gfx1250) — matrix operand order repaired and checked by reading, not compiled, not run on hardware
//
#include <hip/hip_runtime.h>
#include <cmath>

// ---------------- problem constants ----------------
static constexpr int S_    = 2048;
static constexpr int HID_  = 2048;
static constexpr int NH_   = 16;
static constexpr int QLR_  = 1536;
static constexpr int KVLR_ = 512;
static constexpr int QKD_  = 128;
static constexpr int QKP_  = 64;
static constexpr int VD_   = 128;
static constexpr int IH_   = 4;
static constexpr int ID_   = 128;
static constexpr int TOPK_ = 512;
static constexpr int QKDIM_ = KVLR_ + QKP_; // 576

typedef __bf16 bf16x16 __attribute__((ext_vector_type(16)));
typedef float f32x8    __attribute__((ext_vector_type(8)));
typedef int   vi4      __attribute__((vector_size(16)));   // gcc-style, matches builtin

// ---------------- helpers ----------------
__device__ inline __bf16 f2bf(float f) {
  unsigned u = __float_as_uint(f);
  unsigned r = (u + 0x7FFFu + ((u >> 16) & 1u)) >> 16;
  unsigned short h = (unsigned short)r;
  return __builtin_bit_cast(__bf16, h);
}
__device__ inline float bf2f(__bf16 b) {
  unsigned short h = __builtin_bit_cast(unsigned short, b);
  return __uint_as_float(((unsigned)h) << 16);
}

__device__ inline f32x8 wmma_bf16(bf16x16 a, bf16x16 b, f32x8 c) {
  return __builtin_amdgcn_wmma_f32_16x16x32_bf16(false, a, false, b, (short)0, c, false, false);
}

// ---- async global->LDS copy (CDNA5), with safe fallback ----
#if defined(__gfx1250__) && __has_builtin(__builtin_amdgcn_global_load_async_to_lds_b128)
#define ASYNC_LDS 1
#endif

__device__ inline void cp16_async(const __bf16* g, __bf16* l) {
#ifdef ASYNC_LDS
  __builtin_amdgcn_global_load_async_to_lds_b128(
      (__attribute__((address_space(1))) vi4*)(g),
      (__attribute__((address_space(3))) vi4*)(l), 0, 0);
#else
  *(uint4*)l = *(const uint4*)g;
#endif
}
__device__ inline void wait_async() {
#ifdef ASYNC_LDS
#if __has_builtin(__builtin_amdgcn_s_wait_asynccnt)
  __builtin_amdgcn_s_wait_asynccnt(0);
#else
  asm volatile("s_wait_asynccnt 0" ::: "memory");
#endif
#endif
}

// A fragment (16x32 bf16, MxK) from row-major [m][k] LDS.
__device__ inline bf16x16 frag_a(const __bf16* base, int stride, int k0) {
  int lane = threadIdx.x & 31;
  int m  = lane & 15;
  int kh = (lane < 16) ? 0 : 16;
  const __bf16* p = base + m * stride + k0;
  bf16x16 f;
#pragma unroll
  for (int i = 0; i < 16; ++i) f[i] = p[kh + i];
  return f;
}
// B fragment (32x16 bf16, KxN) from LDS stored row-major as [n][k].
__device__ inline bf16x16 frag_b(const __bf16* base, int stride, int k0) {
  int lane = threadIdx.x & 31;
  int n  = lane & 15;
  int kb = (lane < 16) ? 0 : 16;
  const __bf16* p = base + n * stride + k0 + kb;
  bf16x16 f;
#pragma unroll
  for (int i = 0; i < 16; ++i) f[i] = p[i];
  return f;
}
// B fragment from LDS stored as [k][n]
__device__ inline bf16x16 frag_b_kn(const __bf16* base, int rs, int k0) {
  int lane = threadIdx.x & 31;
  int n  = lane & 15;
  int kb = (lane < 16) ? 0 : 16;
  bf16x16 f;
#pragma unroll
  for (int i = 0; i < 16; ++i) f[i] = base[(k0 + kb + i) * rs + n];
  return f;
}

__device__ inline unsigned ordkey(float f) {
  unsigned u = __float_as_uint(f);
  return (u & 0x80000000u) ? ~u : (u | 0x80000000u);
}

__device__ inline void store_c(float* p, float v) { *p = v; }
__device__ inline void store_c(__bf16* p, float v) { *p = f2bf(v); }

// ---------------- bf16 WMMA GEMM: C = A(MxK) * B(NxK)^T ----------------
// Double-buffered LDS, async global->LDS staging. M%128==0, K%32==0; N guarded.
template <typename CT>
__global__ __launch_bounds__(256) void k_gemm2(
    const __bf16* __restrict__ A, long lda, long az,
    const __bf16* __restrict__ Bm, long ldb, long bz,
    CT* __restrict__ C, long ldc, long cz,
    int N, int K)
{
  constexpr int TM = 128, TN = 128, TK = 32;
  __shared__ __align__(16) __bf16 la[2][TM * TK];
  __shared__ __align__(16) __bf16 lb[2][TN * TK];

  int z = blockIdx.z;
  const __bf16* Ap = A + (long)z * az;
  const __bf16* Bp = Bm + (long)z * bz;
  CT* Cp = C + (long)z * cz;

  int m0 = blockIdx.y * TM, n0 = blockIdx.x * TN;
  int tid = threadIdx.x, lane = tid & 31, wid = tid >> 5;
  int wm = (wid & 3) * 32;   // two 16-row m-tiles
  int wn = (wid >> 2) * 64;  // four 16-col n-tiles

  f32x8 acc[2][4] = {};

  auto stage = [&](int buf, int k0) {
#pragma unroll
    for (int i = 0; i < 2; ++i) {
      int c = tid * 2 + i;        // 512 chunks of 16B per operand
      int r = c >> 2, pc = (c & 3) * 8;
      cp16_async(Ap + (long)(m0 + r) * lda + k0 + pc, &la[buf][r * TK + pc]);
      int gn = n0 + r;
      if (gn < N) {
        cp16_async(Bp + (long)gn * ldb + k0 + pc, &lb[buf][r * TK + pc]);
      } else {
        uint4 zz = {0u, 0u, 0u, 0u};
        *(uint4*)&lb[buf][r * TK + pc] = zz;
      }
    }
  };

  stage(0, 0);
  wait_async();
  __syncthreads();

  int nk = K / TK;
  for (int kt = 0; kt < nk; ++kt) {
    int cur = kt & 1;
    if (kt + 1 < nk) stage(cur ^ 1, (kt + 1) * TK);
    bf16x16 a0 = frag_a(&la[cur][wm * TK], TK, 0);
    bf16x16 a1 = frag_a(&la[cur][(wm + 16) * TK], TK, 0);
#pragma unroll
    for (int j = 0; j < 4; ++j) {
      bf16x16 b = frag_b(&lb[cur][(wn + j * 16) * TK], TK, 0);
      acc[0][j] = wmma_bf16(a0, b, acc[0][j]);
      acc[1][j] = wmma_bf16(a1, b, acc[1][j]);
    }
    wait_async();
    __syncthreads();
  }

  int rr = (lane < 16) ? 0 : 8, cn = lane & 15;
#pragma unroll
  for (int i = 0; i < 2; ++i)
#pragma unroll
    for (int j = 0; j < 4; ++j)
#pragma unroll
      for (int r = 0; r < 8; ++r) {
        int gm = m0 + wm + i * 16 + r + rr;
        int gn = n0 + wn + j * 16 + cn;
        if (gn < N) store_c(Cp + (long)gm * ldc + gn, acc[i][j][r]);
      }
}

// ---------------- conversions ----------------
__global__ __launch_bounds__(256) void k_cvt(const float* __restrict__ in, __bf16* __restrict__ out, long n)
{
  long i = (long)blockIdx.x * 256 + threadIdx.x;
  long stride = (long)gridDim.x * 256;
  for (; i < n; i += stride) out[i] = f2bf(in[i]);
}
// wk_up (16,128,512)[h,d,c] -> transposed bf16 [h,c,d]
__global__ __launch_bounds__(256) void k_cvt_wkup(const float* __restrict__ in, __bf16* __restrict__ out)
{
  long gid = (long)blockIdx.x * 256 + threadIdx.x;
  if (gid >= (long)NH_ * KVLR_ * QKD_) return;
  int d = gid & 127;
  int c = (int)((gid >> 7) & 511);
  int h = (int)(gid >> 16);
  out[gid] = f2bf(in[((long)h * QKD_ + d) * KVLR_ + c]);
}

// ---------------- norms ----------------
__global__ __launch_bounds__(256) void k_rmsnorm(
    const float* __restrict__ in, float* __restrict__ out,
    const float* __restrict__ w, int len, long instride, long outstride)
{
  __shared__ float red[256];
  int r = blockIdx.x, tid = threadIdx.x;
  const float* ip = in + (long)r * instride;
  float ss = 0.f;
  for (int i = tid; i < len; i += 256) { float v = ip[i]; ss += v * v; }
  red[tid] = ss; __syncthreads();
  for (int s = 128; s > 0; s >>= 1) { if (tid < s) red[tid] += red[tid + s]; __syncthreads(); }
  float sc = rsqrtf(red[0] / (float)len + 1e-6f);
  float* op = out + (long)r * outstride;
  for (int i = tid; i < len; i += 256) op[i] = ip[i] * sc * w[i];
}

__global__ __launch_bounds__(128) void k_layernorm128(
    float* __restrict__ x, const float* __restrict__ w, const float* __restrict__ b)
{
  __shared__ float red[128];
  int r = blockIdx.x, tid = threadIdx.x;
  float* p = x + (long)r * 128;
  float v = p[tid];
  red[tid] = v; __syncthreads();
  for (int s = 64; s > 0; s >>= 1) { if (tid < s) red[tid] += red[tid + s]; __syncthreads(); }
  float mean = red[0] / 128.f; __syncthreads();
  float d = v - mean;
  red[tid] = d * d; __syncthreads();
  for (int s = 64; s > 0; s >>= 1) { if (tid < s) red[tid] += red[tid + s]; __syncthreads(); }
  float var = red[0] / 128.f;
  p[tid] = d * rsqrtf(var + 1e-6f) * w[tid] + b[tid];
}

// ---------------- RoPE ----------------
__device__ inline void rope_cs(int s, int p, float& c, float& si) {
  float inv = __expf(-(float)(2 * p) * (9.210340371976184f / 64.0f)); // 10000^(-2p/64)
  float ang = (float)s * inv;
  c = __cosf(ang); si = __sinf(ang);
}

__global__ __launch_bounds__(256) void k_rope_kpe(const float* __restrict__ kv, float* __restrict__ keys)
{
  int gid = blockIdx.x * 256 + threadIdx.x;
  if (gid >= S_ * 32) return;
  int s = gid >> 5, p = gid & 31;
  float c, si; rope_cs(s, p, c, si);
  float x1 = kv[(long)s * 576 + 512 + 2 * p];
  float x2 = kv[(long)s * 576 + 512 + 2 * p + 1];
  keys[(long)s * 576 + 512 + 2 * p]     = x1 * c - x2 * si;
  keys[(long)s * 576 + 512 + 2 * p + 1] = x1 * si + x2 * c;
}

__global__ __launch_bounds__(256) void k_rope_qpe(const __bf16* __restrict__ q, __bf16* __restrict__ qabs)
{
  int gid = blockIdx.x * 256 + threadIdx.x;
  if (gid >= S_ * NH_ * 32) return;
  int p = gid & 31, h = (gid >> 5) & 15, s = gid >> 9;
  float c, si; rope_cs(s, p, c, si);
  const __bf16* src = q + (long)s * 3072 + h * 192 + 128;
  float x1 = bf2f(src[2 * p]), x2 = bf2f(src[2 * p + 1]);
  __bf16* dst = qabs + (long)s * (NH_ * QKDIM_) + h * QKDIM_ + 512;
  dst[2 * p]     = f2bf(x1 * c - x2 * si);
  dst[2 * p + 1] = f2bf(x1 * si + x2 * c);
}

// half-rope in place: rows = s*H + h, row length 128, seq pos = s
__global__ __launch_bounds__(256) void k_rope_half_hs(float* __restrict__ x, int H)
{
  int gid = blockIdx.x * 256 + threadIdx.x;
  if (gid >= S_ * H * 32) return;
  int p = gid & 31;
  int row = gid >> 5;
  int s = row / H;
  float c, si; rope_cs(s, p, c, si);
  float* base = x + (long)row * 128;
  float x1 = base[p], x2 = base[32 + p];
  base[p]      = x1 * c - x2 * si;
  base[32 + p] = x1 * si + x2 * c;
}

__global__ void k_zero1(float* p) { if (threadIdx.x == 0 && blockIdx.x == 0) *p = 0.f; }

// iw[s,h] = x[s,:] . idx_ww[h,:]
__global__ __launch_bounds__(256) void k_iw(const float* __restrict__ x, const float* __restrict__ ww,
                                            float* __restrict__ iw)
{
  int gid = blockIdx.x * 256 + threadIdx.x;
  if (gid >= S_ * IH_) return;
  int h = gid & 3, s = gid >> 2;
  const float* xr = x + (long)s * HID_;
  const float* wr = ww + (long)h * HID_;
  float d = 0.f;
  for (int i = 0; i < HID_; ++i) d += xr[i] * wr[i];
  iw[gid] = d;
}

// ---------------- indexer: logits + top-k + log-softmax ----------------
__global__ __launch_bounds__(256) void k_indexer(
    const float* __restrict__ iq, const float* __restrict__ ik, const float* __restrict__ iw,
    int* __restrict__ tki, float* __restrict__ lp, float idx_scale, float w_scale)
{
  __shared__ float lg[S_];
  __shared__ float qloc[IH_ * ID_];
  __shared__ float wloc[IH_];
  __shared__ unsigned hist[256];
  __shared__ float vals[TOPK_];
  __shared__ float red[256];
  __shared__ unsigned sh_prefix, sh_rem, c_gt, c_eq;

  int t = blockIdx.x, tid = threadIdx.x;
  for (int e = tid; e < IH_ * ID_; e += 256) qloc[e] = iq[(long)t * (IH_ * ID_) + e];
  if (tid < IH_) wloc[tid] = iw[(long)t * IH_ + tid] * w_scale;
  __syncthreads();

  for (int s = tid; s < S_; s += 256) {
    const float* kr = ik + (long)s * ID_;
    float d0 = 0.f, d1 = 0.f, d2 = 0.f, d3 = 0.f;
    for (int i = 0; i < ID_; ++i) {
      float kvv = kr[i];
      d0 += qloc[i] * kvv;        d1 += qloc[128 + i] * kvv;
      d2 += qloc[256 + i] * kvv;  d3 += qloc[384 + i] * kvv;
    }
    float acc = wloc[0] * fmaxf(d0 * idx_scale, 0.f) + wloc[1] * fmaxf(d1 * idx_scale, 0.f)
              + wloc[2] * fmaxf(d2 * idx_scale, 0.f) + wloc[3] * fmaxf(d3 * idx_scale, 0.f);
    lg[s] = acc + ((s <= t) ? 0.f : -1.0e9f);
  }
  __syncthreads();

  unsigned prefix = 0, rem = TOPK_;
  for (int pass = 0; pass < 4; ++pass) {
    int shift = 24 - 8 * pass;
    for (int b = tid; b < 256; b += 256) hist[b] = 0;
    __syncthreads();
    for (int s = tid; s < S_; s += 256) {
      unsigned u = ordkey(lg[s]);
      bool ok = (pass == 0) || ((u >> (shift + 8)) == (prefix >> (shift + 8)));
      if (ok) atomicAdd(&hist[(u >> shift) & 255u], 1u);
    }
    __syncthreads();
    if (tid == 0) {
      unsigned acc = 0; int sel = 0; unsigned above = 0;
      for (int b = 255; b >= 0; --b) {
        if (acc + hist[b] >= rem) { sel = b; above = acc; break; }
        acc += hist[b];
      }
      sh_prefix = prefix | ((unsigned)sel << shift);
      sh_rem = rem - above;
    }
    __syncthreads();
    prefix = sh_prefix; rem = sh_rem;
    __syncthreads();
  }

  if (tid == 0) { c_gt = 0; c_eq = 0; }
  __syncthreads();
  unsigned T = prefix;
  unsigned n_gt = TOPK_ - rem;
  for (int s = tid; s < S_; s += 256) {
    unsigned u = ordkey(lg[s]);
    int slot = -1;
    if (u > T) slot = (int)atomicAdd(&c_gt, 1u);
    else if (u == T) { unsigned e = atomicAdd(&c_eq, 1u); if (e < rem) slot = (int)(n_gt + e); }
    if (slot >= 0) { tki[(long)t * TOPK_ + slot] = s; vals[slot] = lg[s]; }
  }
  __syncthreads();

  float mv = -3.0e38f;
  for (int j = tid; j < TOPK_; j += 256) mv = fmaxf(mv, vals[j]);
  red[tid] = mv; __syncthreads();
  for (int s2 = 128; s2 > 0; s2 >>= 1) { if (tid < s2) red[tid] = fmaxf(red[tid], red[tid + s2]); __syncthreads(); }
  mv = red[0]; __syncthreads();
  float se = 0.f;
  for (int j = tid; j < TOPK_; j += 256) se += __expf(vals[j] - mv);
  red[tid] = se; __syncthreads();
  for (int s2 = 128; s2 > 0; s2 >>= 1) { if (tid < s2) red[tid] += red[tid + s2]; __syncthreads(); }
  float lse = logf(red[0]);
  for (int j = tid; j < TOPK_; j += 256) lp[(long)t * TOPK_ + j] = vals[j] - mv - lse;
}

// ---------------- fused sparse attention ----------------
__global__ __launch_bounds__(256) void k_attn(
    const __bf16* __restrict__ qabsb,  // (S,16,576) bf16
    const __bf16* __restrict__ keysb,  // (S,576) bf16
    const int*   __restrict__ tki,
    const float* __restrict__ lp,
    __bf16* __restrict__ attn_out,     // (S,16,512) bf16
    float* __restrict__ kl, float scale)
{
  __shared__ uint4 qb4[16 * 72];       // 16x576 bf16
  __shared__ uint4 kb4[64 * 72];       // 64x576 bf16
  __shared__ float  sf[16 * 512];
  __shared__ __align__(16) __bf16 pb[16 * 64];
  __shared__ int    idx_all[512];
  __shared__ float  cls[512];
  __shared__ float  red[256];
  const __bf16* qb = (const __bf16*)qb4;
  __bf16* kb = (__bf16*)kb4;

  int t = blockIdx.x;
  int tid = threadIdx.x, lane = tid & 31, wid = tid >> 5;

  {
    const __bf16* qs = qabsb + (long)t * 9216;
    for (int e = tid; e < 16 * 72; e += 256) cp16_async(qs + e * 8, (__bf16*)&qb4[e]);
  }
  for (int j = tid; j < 512; j += 256) idx_all[j] = tki[(long)t * 512 + j];
  for (int e = tid; e < 16 * 512; e += 256) sf[e] = 0.f;
  wait_async();
  __syncthreads();

  // Phase A: scores
  for (int c = 0; c < 8; ++c) {
    for (int e = tid; e < 64 * 72; e += 256) {
      int j = e / 72, ci = e - j * 72;
      cp16_async(keysb + (long)idx_all[c * 64 + j] * 576 + ci * 8, (__bf16*)&kb4[e]);
    }
    wait_async();
    __syncthreads();
    int ntile = wid & 3;
    int khalf = wid >> 2;
    f32x8 acc = {};
    for (int ks = khalf * 9; ks < khalf * 9 + 9; ++ks) {
      bf16x16 a = frag_a(qb, 576, ks * 32);
      bf16x16 b = frag_b(kb + ntile * 16 * 576, 576, ks * 32);
      acc = wmma_bf16(a, b, acc);
    }
    int rr = (lane < 16) ? 0 : 8, cn = lane & 15;
#pragma unroll
    for (int r = 0; r < 8; ++r)
      atomicAdd(&sf[(r + rr) * 512 + c * 64 + ntile * 16 + cn], acc[r] * scale);
    __syncthreads();
  }

  // Phase B: per-head softmax (+ causal mask)
  for (int h = wid; h < 16; h += 8) {
    float mv = -3.0e38f;
    for (int j = lane; j < 512; j += 32) {
      float s = sf[h * 512 + j] + ((idx_all[j] <= t) ? 0.f : -1.0e9f);
      mv = fmaxf(mv, s);
    }
    for (int o = 16; o >= 1; o >>= 1) mv = fmaxf(mv, __shfl_xor(mv, o));
    float lsum = 0.f;
    for (int j = lane; j < 512; j += 32) {
      float s = sf[h * 512 + j] + ((idx_all[j] <= t) ? 0.f : -1.0e9f);
      lsum += __expf(s - mv);
    }
    for (int o = 16; o >= 1; o >>= 1) lsum += __shfl_xor(lsum, o);
    float inv = 1.0f / lsum;
    for (int j = lane; j < 512; j += 32) {
      float s = sf[h * 512 + j] + ((idx_all[j] <= t) ? 0.f : -1.0e9f);
      sf[h * 512 + j] = __expf(s - mv) * inv;
    }
  }
  __syncthreads();

  // KL contribution
  for (int j = tid; j < 512; j += 256) {
    float cs = 0.f;
    for (int h = 0; h < 16; ++h) cs += sf[h * 512 + j];
    cls[j] = cs;
  }
  __syncthreads();
  float tt = 0.f;
  for (int j = tid; j < 512; j += 256) tt += cls[j];
  red[tid] = tt; __syncthreads();
  for (int s2 = 128; s2 > 0; s2 >>= 1) { if (tid < s2) red[tid] += red[tid + s2]; __syncthreads(); }
  float total = red[0];
  __syncthreads();
  float klloc = 0.f;
  for (int j = tid; j < 512; j += 256) {
    float as = cls[j] / total;
    float tg = logf(as); tg = fminf(fmaxf(tg, -100.f), 0.f);
    float ip = lp[(long)t * 512 + j]; ip = fminf(fmaxf(ip, -100.f), 0.f);
    klloc += __expf(tg) * (tg - ip);
  }
  red[tid] = klloc; __syncthreads();
  for (int s2 = 128; s2 > 0; s2 >>= 1) { if (tid < s2) red[tid] += red[tid + s2]; __syncthreads(); }
  if (tid == 0) atomicAdd(kl, red[0]);
  __syncthreads();

  // Phase C: attn = P(16x512) . V(512x512), V = kv_c rows of gathered keys
  f32x8 o0 = {}, o1 = {}, o2 = {}, o3 = {};
  int d0 = wid * 64;
  for (int c = 0; c < 8; ++c) {
    for (int e = tid; e < 64 * 64; e += 256) {
      int j = e >> 6, ci = e & 63;
      cp16_async(keysb + (long)idx_all[c * 64 + j] * 576 + ci * 8, (__bf16*)&kb4[j * 64 + ci]);
    }
    for (int e = tid; e < 16 * 64; e += 256) {
      int h = e >> 6, j = e & 63;
      pb[e] = f2bf(sf[h * 512 + c * 64 + j]);
    }
    wait_async();
    __syncthreads();
#pragma unroll
    for (int ks = 0; ks < 2; ++ks) {
      bf16x16 a = frag_a(pb, 64, ks * 32);
      o0 = wmma_bf16(a, frag_b_kn(kb + d0 +  0, 512, ks * 32), o0);
      o1 = wmma_bf16(a, frag_b_kn(kb + d0 + 16, 512, ks * 32), o1);
      o2 = wmma_bf16(a, frag_b_kn(kb + d0 + 32, 512, ks * 32), o2);
      o3 = wmma_bf16(a, frag_b_kn(kb + d0 + 48, 512, ks * 32), o3);
    }
    __syncthreads();
  }
  int rr = (lane < 16) ? 0 : 8, cn = lane & 15;
#pragma unroll
  for (int r = 0; r < 8; ++r) {
    int h = r + rr;
    long base = (long)t * 8192 + (long)h * 512 + d0;
    attn_out[base +  0 + cn] = f2bf(o0[r]);
    attn_out[base + 16 + cn] = f2bf(o1[r]);
    attn_out[base + 32 + cn] = f2bf(o2[r]);
    attn_out[base + 48 + cn] = f2bf(o3[r]);
  }
}

// ---------------- host ----------------
template <typename CT>
static void gemm2(hipStream_t st, const __bf16* A, long lda, long az,
                  const __bf16* B, long ldb, long bz,
                  CT* C, long ldc, long cz, int M, int N, int K, int Z)
{
  dim3 g((N + 127) / 128, M / 128, Z);
  k_gemm2<CT><<<g, 256, 0, st>>>(A, lda, az, B, ldb, bz, C, ldc, cz, N, K);
}
static void cvt(hipStream_t st, const float* in, __bf16* out, long n)
{
  int blocks = (int)((n + 255 * 256) / (256 * 256));
  if (blocks < 1) blocks = 1;
  if (blocks > 4096) blocks = 4096;
  k_cvt<<<blocks, 256, 0, st>>>(in, out, n);
}

extern "C" void kernel_launch(void* const* d_in, const int* in_sizes, int n_in,
                              void* d_out, int out_size, void* d_ws, size_t ws_size,
                              hipStream_t stream)
{
  (void)in_sizes; (void)n_in; (void)out_size; (void)ws_size;
  const float* x        = (const float*)d_in[0];
  const float* wq_down  = (const float*)d_in[2];
  const float* q_ln_w   = (const float*)d_in[3];
  const float* wq_up    = (const float*)d_in[4];
  const float* wkv_down = (const float*)d_in[5];
  const float* kv_ln_w  = (const float*)d_in[6];
  const float* wk_up    = (const float*)d_in[7];
  const float* wv_up    = (const float*)d_in[8];
  const float* wo       = (const float*)d_in[9];
  const float* idx_wq_b = (const float*)d_in[10];
  const float* idx_wk   = (const float*)d_in[11];
  const float* idx_kn_w = (const float*)d_in[12];
  const float* idx_kn_b = (const float*)d_in[13];
  const float* idx_ww   = (const float*)d_in[14];

  float* out = (float*)d_out;
  float* kl  = out + (size_t)S_ * HID_;

  char* wsb = (char*)d_ws;
  auto alloc = [&](size_t bytes) -> void* {
    void* r = (void*)wsb;
    wsb += (bytes + 255) & ~(size_t)255;
    return r;
  };
  // fp32 intermediates
  float* qr   = (float*)alloc((size_t)S_ * QLR_ * 4);
  float* kv   = (float*)alloc((size_t)S_ * QKDIM_ * 4);
  float* keys = (float*)alloc((size_t)S_ * QKDIM_ * 4);
  float* iq   = (float*)alloc((size_t)S_ * IH_ * ID_ * 4);
  float* ik   = (float*)alloc((size_t)S_ * ID_ * 4);
  float* iw   = (float*)alloc((size_t)S_ * IH_ * 4);
  float* lp   = (float*)alloc((size_t)S_ * TOPK_ * 4);
  int*   tki  = (int*)alloc((size_t)S_ * TOPK_ * 4);
  // bf16 tensors
  __bf16* x_bf       = (__bf16*)alloc((size_t)S_ * HID_ * 2);
  __bf16* qr_bf      = (__bf16*)alloc((size_t)S_ * QLR_ * 2);
  __bf16* q_bf       = (__bf16*)alloc((size_t)S_ * NH_ * 192 * 2);
  __bf16* qabs_bf    = (__bf16*)alloc((size_t)S_ * NH_ * QKDIM_ * 2);
  __bf16* keys_bf    = (__bf16*)alloc((size_t)S_ * QKDIM_ * 2);
  __bf16* attn_bf    = (__bf16*)alloc((size_t)S_ * NH_ * KVLR_ * 2);
  __bf16* attnup_bf  = (__bf16*)alloc((size_t)S_ * NH_ * VD_ * 2);
  // bf16 weights
  __bf16* wqd_bf  = (__bf16*)alloc((size_t)QLR_ * HID_ * 2);
  __bf16* wqu_bf  = (__bf16*)alloc((size_t)NH_ * 192 * QLR_ * 2);
  __bf16* wkvd_bf = (__bf16*)alloc((size_t)QKDIM_ * HID_ * 2);
  __bf16* wku_bf  = (__bf16*)alloc((size_t)NH_ * KVLR_ * QKD_ * 2);   // transposed [h,c,d]
  __bf16* wvu_bf  = (__bf16*)alloc((size_t)NH_ * VD_ * KVLR_ * 2);
  __bf16* wo_bf   = (__bf16*)alloc((size_t)HID_ * NH_ * VD_ * 2);
  __bf16* iwq_bf  = (__bf16*)alloc((size_t)IH_ * ID_ * QLR_ * 2);
  __bf16* iwk_bf  = (__bf16*)alloc((size_t)ID_ * HID_ * 2);

  const double ms = 0.1 * 1.0 * log(40.0) + 1.0;
  const float SCALE = (float)(ms * ms / sqrt((double)(QKD_ + QKP_)));
  const float IDXS  = (float)(1.0 / sqrt((double)ID_));
  const float WSC   = (float)(1.0 / sqrt((double)IH_));

  // bf16 conversions of inputs
  cvt(stream, x, x_bf, (long)S_ * HID_);
  cvt(stream, wq_down, wqd_bf, (long)QLR_ * HID_);
  cvt(stream, wq_up, wqu_bf, (long)NH_ * 192 * QLR_);
  cvt(stream, wkv_down, wkvd_bf, (long)QKDIM_ * HID_);
  cvt(stream, wv_up, wvu_bf, (long)NH_ * VD_ * KVLR_);
  cvt(stream, wo, wo_bf, (long)HID_ * NH_ * VD_);
  cvt(stream, idx_wq_b, iwq_bf, (long)IH_ * ID_ * QLR_);
  cvt(stream, idx_wk, iwk_bf, (long)ID_ * HID_);
  k_cvt_wkup<<<((long)NH_ * KVLR_ * QKD_ + 255) / 256, 256, 0, stream>>>(wk_up, wku_bf);

  // 1) qr = rmsnorm(x @ wq_down^T)
  gemm2<float>(stream, x_bf, HID_, 0, wqd_bf, HID_, 0, qr, QLR_, 0, S_, QLR_, HID_, 1);
  k_rmsnorm<<<S_, 256, 0, stream>>>(qr, qr, q_ln_w, QLR_, QLR_, QLR_);
  cvt(stream, qr, qr_bf, (long)S_ * QLR_);
  // 2) q = qr @ wq_up^T (bf16 out)
  gemm2<__bf16>(stream, qr_bf, QLR_, 0, wqu_bf, QLR_, 0, q_bf, NH_ * 192, 0, S_, NH_ * 192, QLR_, 1);
  // 3) kv = x @ wkv_down^T
  gemm2<float>(stream, x_bf, HID_, 0, wkvd_bf, HID_, 0, kv, QKDIM_, 0, S_, QKDIM_, HID_, 1);
  // 4) keys = [rmsnorm(kv_c) | rope(k_pe)]
  k_rmsnorm<<<S_, 256, 0, stream>>>(kv, keys, kv_ln_w, KVLR_, QKDIM_, QKDIM_);
  k_rope_kpe<<<(S_ * 32 + 255) / 256, 256, 0, stream>>>(kv, keys);
  cvt(stream, keys, keys_bf, (long)S_ * QKDIM_);
  // 5) q_abs[:, :, :512] = q_nope @ k_up_w[h]^T  (bf16 out, batched over heads)
  gemm2<__bf16>(stream, q_bf, NH_ * 192, 192, wku_bf, QKD_, (long)KVLR_ * QKD_,
                qabs_bf, NH_ * QKDIM_, QKDIM_, S_, KVLR_, QKD_, NH_);
  // 6) q_abs[:, :, 512:] = rope(q_pe)
  k_rope_qpe<<<(S_ * NH_ * 32 + 255) / 256, 256, 0, stream>>>(q_bf, qabs_bf);
  // 7) indexer projections
  gemm2<float>(stream, qr_bf, QLR_, 0, iwq_bf, QLR_, 0, iq, IH_ * ID_, 0, S_, IH_ * ID_, QLR_, 1);
  gemm2<float>(stream, x_bf, HID_, 0, iwk_bf, HID_, 0, ik, ID_, 0, S_, ID_, HID_, 1);
  k_iw<<<(S_ * IH_ + 255) / 256, 256, 0, stream>>>(x, idx_ww, iw);
  k_layernorm128<<<S_, 128, 0, stream>>>(ik, idx_kn_w, idx_kn_b);
  k_rope_half_hs<<<(S_ * 32 + 255) / 256, 256, 0, stream>>>(ik, 1);
  k_rope_half_hs<<<(S_ * IH_ * 32 + 255) / 256, 256, 0, stream>>>(iq, IH_);
  // 8) indexer logits + top-k + log-softmax
  k_indexer<<<S_, 256, 0, stream>>>(iq, ik, iw, tki, lp, IDXS, WSC);
  // 9) sparse attention + KL
  k_zero1<<<1, 1, 0, stream>>>(kl);
  k_attn<<<S_, 256, 0, stream>>>(qabs_bf, keys_bf, tki, lp, attn_bf, kl, SCALE);
  // 10) per-head V up-projection (bf16 out)
  gemm2<__bf16>(stream, attn_bf, NH_ * KVLR_, KVLR_, wvu_bf, KVLR_, (long)VD_ * KVLR_,
                attnup_bf, NH_ * VD_, VD_, S_, VD_, KVLR_, NH_);
  // 11) out = attn_up @ wo^T
  gemm2<float>(stream, attnup_bf, NH_ * VD_, 0, wo_bf, NH_ * VD_, 0, out, HID_, 0, S_, HID_, NH_ * VD_, 1);
}
